// KAN_DiffPhys_81415400063379
// MI455X (gfx1250) — matrix operand order repaired and checked by reading, not compiled, not run on hardware
//
#include <hip/hip_runtime.h>
#include <hip/hip_bf16.h>

#define T_STEPS 100000
#define GRID_N  30
#define HIDDEN  16
#define TILES   (T_STEPS / 16)   // 6250 tiles of 16 rows
#define WAVES_PER_BLOCK 8

typedef __attribute__((ext_vector_type(16))) _Float16 v16h;
typedef __attribute__((ext_vector_type(8)))  float    v8f;

__device__ __forceinline__ float softplus_f(float x) {
    return (x > 20.0f) ? x : log1pf(__expf(x));
}

// ---------------------------------------------------------------------------
// Kernel A: per 16-row tile, one wave computes
//   h = WMMA( RBF_basis(t) , spline_weight1 )   (f16 in, f32 acc)
//   beta_dt = softplus( layer2(h) ) * dt
// ---------------------------------------------------------------------------
__global__ __launch_bounds__(256) void kan_beta_kernel(
    const float* __restrict__ t,     // [T]
    const float* __restrict__ sw1,   // [30,16] row-major
    const float* __restrict__ bw1,   // [16]
    const float* __restrict__ sw2,   // [480] = [16*30]
    const float* __restrict__ bw2,   // [16]
    float* __restrict__ bdt_out)     // [T]
{
    __shared__ float s_sw2[HIDDEN * GRID_N];              // 480 floats
    __shared__ float s_h[WAVES_PER_BLOCK][16 * 20];       // padded 16x16 tiles

    // block-cooperative preload of spline_weight2 (re-read 30x per element)
    for (int k = threadIdx.x; k < HIDDEN * GRID_N; k += blockDim.x)
        s_sw2[k] = sw2[k];
    __syncthreads();

    const int lane = threadIdx.x & 31;
    const int wave = threadIdx.x >> 5;
    const int tile = blockIdx.x * WAVES_PER_BLOCK + wave;
    if (tile >= TILES) return;          // wave-uniform exit: EXEC all-1s for WMMA

    const bool hi  = lane >= 16;
    const int  nl  = lane & 15;
    const float dt    = 1.0f / (float)T_STEPS;
    const float inv29 = 1.0f / 29.0f;   // linspace(0,1,30) step

    // t for this lane's A-matrix row (row M = nl for both lane halves)
    const float tr = t[tile * 16 + nl];

    // ---- A fragment: 16x32 f16 basis values, K padded 30->32 with zeros ----
    // lanes 0-15 : elements 0-7 -> K=0..7,  elements 8-15 -> K=16..23
    // lanes 16-31: elements 0-7 -> K=8..15, elements 8-15 -> K=24..31
    v16h a;
    #pragma unroll
    for (int j = 0; j < 16; ++j) {
        int K = j + (j < 8 ? 0 : 8) + (hi ? 8 : 0);
        float g = (float)K * inv29;
        float d = tr - g;
        float val = (K < GRID_N) ? __expf(-20.0f * d * d) : 0.0f;
        a[j] = (_Float16)val;
    }

    // ---- B fragment: 32x16 f16 = spline_weight1 padded; col N = nl ----
    // lanes 0-15 : elements 0-7 -> K=0..7,  elements 8-15 -> K=16..23; lanes 16-31: K=8..15, K=24..31
    v16h b;
    #pragma unroll
    for (int j = 0; j < 16; ++j) {
        int K = j + (j < 8 ? 0 : 8) + (hi ? 8 : 0);
        float val = (K < GRID_N) ? sw1[K * HIDDEN + nl] : 0.0f;
        b[j] = (_Float16)val;
    }

    v8f c = {};
    c = __builtin_amdgcn_wmma_f32_16x16x32_f16(
            /*neg_a=*/false, a, /*neg_b=*/false, b,
            /*c_mod=*/(short)0, c, /*reuse_a=*/false, /*reuse_b=*/false);

    // ---- stage D tile through LDS to transpose (lane=col -> lane=row) ----
    #pragma unroll
    for (int v = 0; v < 8; ++v) {
        int M = v + (hi ? 8 : 0);
        s_h[wave][M * 20 + nl] = c[v];
    }
    // same-wave LDS store->load ordering; stop compiler reordering too
    asm volatile("s_wait_dscnt 0" ::: "memory");

    // ---- layer 2: lane handles row r = nl, hidden slice [hi*8, hi*8+8) ----
    const int r = nl;
    float acc = 0.0f;
    #pragma unroll
    for (int ii = 0; ii < 8; ++ii) {
        int i = (hi ? 8 : 0) + ii;
        // add base term t*W1_base here (C was zero in the WMMA)
        float hv = s_h[wave][r * 20 + i] + tr * bw1[i];
        acc += hv * bw2[i];                       // layer-2 base path
        #pragma unroll
        for (int g = 0; g < GRID_N; ++g) {
            float d = hv - (float)g * inv29;
            acc += __expf(-20.0f * d * d) * s_sw2[i * GRID_N + g];
        }
    }
    acc += __shfl_xor(acc, 16, 32);               // combine the two half-sums
    if (!hi) {
        bdt_out[tile * 16 + r] = softplus_f(acc) * dt;
    }
}

// ---------------------------------------------------------------------------
// Kernel B: sequential SIR recurrence (nonlinear in S*I -> no parallel scan).
// Minimal critical path per step: p = bdt*S*I; S = sat(S-p); I = sat(I*k + p)
// float4 load/store amortizes memory ops over 4 steps.
// ---------------------------------------------------------------------------
__global__ void sir_scan_kernel(const float* __restrict__ bdt,
                                const float* __restrict__ I_init,
                                const float* __restrict__ gamma_param,
                                float* __restrict__ out)
{
    if (threadIdx.x != 0) return;
    const float dt    = 1.0f / (float)T_STEPS;
    const float gamma = softplus_f(gamma_param[0]);
    const float k1mg  = 1.0f - gamma * dt;        // I*(1-gamma*dt) + beta*dt*S*I

    float I = I_init[0];
    float S = 1.0f - I;

    const float4* b4 = (const float4*)bdt;
    float4*       o4 = (float4*)out;

    for (int i = 0; i < T_STEPS / 4; ++i) {
        float4 bb = b4[i];
        float4 oo;
        { float p = bb.x * S * I; S = __saturatef(S - p); I = __saturatef(I * k1mg + p); oo.x = I; }
        { float p = bb.y * S * I; S = __saturatef(S - p); I = __saturatef(I * k1mg + p); oo.y = I; }
        { float p = bb.z * S * I; S = __saturatef(S - p); I = __saturatef(I * k1mg + p); oo.z = I; }
        { float p = bb.w * S * I; S = __saturatef(S - p); I = __saturatef(I * k1mg + p); oo.w = I; }
        o4[i] = oo;
    }
}

// ---------------------------------------------------------------------------
extern "C" void kernel_launch(void* const* d_in, const int* in_sizes, int n_in,
                              void* d_out, int out_size, void* d_ws, size_t ws_size,
                              hipStream_t stream) {
    const float* t_arr = (const float*)d_in[0];   // [100000]
    const float* I0    = (const float*)d_in[1];   // [1]
    const float* sw1   = (const float*)d_in[2];   // [30*16]
    const float* bw1   = (const float*)d_in[3];   // [16]
    const float* sw2   = (const float*)d_in[4];   // [480]
    const float* bw2   = (const float*)d_in[5];   // [16]
    const float* gp    = (const float*)d_in[6];   // [1]
    float* out = (float*)d_out;
    float* bdt = (float*)d_ws;                    // 400 KB scratch: beta*dt

    dim3 grid((TILES + WAVES_PER_BLOCK - 1) / WAVES_PER_BLOCK);  // 782 blocks
    kan_beta_kernel<<<grid, 256, 0, stream>>>(t_arr, sw1, bw1, sw2, bw2, bdt);
    sir_scan_kernel<<<1, 32, 0, stream>>>(bdt, I0, gp, out);
}
